// LoRAMLP_86595130622105
// MI455X (gfx1250) — matrix operand order repaired and checked by reading, not compiled, not run on hardware
//
#include <hip/hip_runtime.h>
#include <math.h>

// ---------------------------------------------------------------------------
// LoRA MLP for MI455X (gfx1250, wave32, WMMA).
//   gate_up = x @ W_gu + (x @ A_gu) @ B_gu ; h = silu(gate)*up
//   out     = h @ W_d  + (h @ A_d ) @ B_d
// Strategy:
//  * one-shot fp32->bf16 conversion pass for x / W_gu / W_d (bandwidth-bound,
//    halves L2 footprint so x+W stay L2-resident across N-tiles)
//  * bf16 WMMA (v_wmma_f32_16x16x32_bf16) everywhere; f32 accumulate
//  * h kept in bf16 scratch (halves h HBM traffic)
//  * double-buffered LDS pipeline, ONE barrier per K-step; X tiles staged by
//    GLOBAL_LOAD_ASYNC_TO_LDS_B128 (ASYNCcnt), W tiles by b128 load +
//    transpose ds_store; zero conversion VALU in the K-loop
// ---------------------------------------------------------------------------

#define TOKENS  16384
#define D_MODEL 1024
#define D_FF    2816
#define RANK    16

typedef __attribute__((ext_vector_type(16))) __bf16 v16bf;
typedef __attribute__((ext_vector_type(2)))  __bf16 v2bf;
typedef __attribute__((ext_vector_type(8)))  float  v8f;

union Frag16 {              // one WMMA 16-bit A/B operand: 8 VGPRs
  uint4 q[2];
  v16bf v;
};

// fp32 -> bf16 (RNE) via native fptrunc; backend picks v_cvt_*_bf16 on gfx1250
__device__ __forceinline__ unsigned bfpack2(float a, float b) {
  v2bf p = {(__bf16)a, (__bf16)b};
  return __builtin_bit_cast(unsigned, p);
}
__device__ __forceinline__ unsigned short bf1(float a) {
  __bf16 h = (__bf16)a;
  return __builtin_bit_cast(unsigned short, h);
}
__device__ __forceinline__ float bf2f(unsigned short s) {
  return __uint_as_float(((unsigned)s) << 16);
}

// ---------------------------------------------------------------------------
// One-shot fp32 -> bf16 conversion (n must be a multiple of 2048).
// ---------------------------------------------------------------------------
__global__ void __launch_bounds__(256)
cvt_f32_to_bf16(const float* __restrict__ src, unsigned short* __restrict__ dst,
                long n) {
  long i = ((long)blockIdx.x * 256 + threadIdx.x) * 8;
  if (i + 8 > n) return;
  float4 f0 = reinterpret_cast<const float4*>(src + i)[0];
  float4 f1 = reinterpret_cast<const float4*>(src + i)[1];
  uint4 u = make_uint4(bfpack2(f0.x, f0.y), bfpack2(f0.z, f0.w),
                       bfpack2(f1.x, f1.y), bfpack2(f1.z, f1.w));
  reinterpret_cast<uint4*>(dst + i)[0] = u;
}

// ---------------------------------------------------------------------------
// Rank-16 projections (XA = X @ A).  ~0.2% of total FLOPs -> simple VALU.
// ---------------------------------------------------------------------------
__global__ void __launch_bounds__(256)
rank_proj_f32(const float* __restrict__ X, const float* __restrict__ A,
              float* __restrict__ out, int K) {
  const int t = blockIdx.x * 16 + (threadIdx.x >> 4);
  const int r = threadIdx.x & 15;
  const float* xr = X + (size_t)t * K;
  float acc = 0.f;
  for (int k = 0; k < K; ++k) acc = fmaf(xr[k], A[(size_t)k * RANK + r], acc);
  out[(size_t)t * RANK + r] = acc;
}

__global__ void __launch_bounds__(256)
rank_proj_bf16(const unsigned short* __restrict__ X, const float* __restrict__ A,
               float* __restrict__ out, int K) {
  const int t = blockIdx.x * 16 + (threadIdx.x >> 4);
  const int r = threadIdx.x & 15;
  const unsigned short* xr = X + (size_t)t * K;
  float acc = 0.f;
  for (int k = 0; k < K; ++k) acc = fmaf(bf2f(xr[k]), A[(size_t)k * RANK + r], acc);
  out[(size_t)t * RANK + r] = acc;
}

// ---------------------------------------------------------------------------
// Fused tiled GEMM + rank-16 LoRA (+ optional SiLU-and-mul when GATED).
//   block tile: M=128 x N=64, 256 threads = 8 waves in a 4(M) x 2(N) grid.
//   Each wave owns 2x2 (GATED: gate+up => 8) 16x16 f32 accumulators.
//   All operands bf16.  K staged 32 at a time, DOUBLE-BUFFERED:
//     issue(next: X async->LDS, W b128) -> wmma(cur) -> commit(next: W
//     transpose ds_store) -> s_wait_asynccnt -> barrier
// ---------------------------------------------------------------------------
#define XS_STRIDE 40   // shorts per x-tile row (32 + pad, keeps 16B alignment)
#define WT_STRIDE 40   // shorts per transposed-W row

template <int GATED>
__global__ void __launch_bounds__(256)
fused_lora_gemm(const unsigned short* __restrict__ Xb,   // [T, K] bf16
                const unsigned short* __restrict__ Wb,   // [K, NW] bf16
                const float* __restrict__ XA,            // [T, 16] f32
                const float* __restrict__ BL,            // [16, NW] f32
                void* __restrict__ Out_, int K, int NW, int NOUT) {
  __shared__ unsigned short xs[2][128 * XS_STRIDE];
  __shared__ unsigned short wgT[2][64 * WT_STRIDE];
  __shared__ unsigned short wuT[2][64 * WT_STRIDE];

  const int tid  = threadIdx.x;
  const int lane = tid & 31;
  const int wid  = tid >> 5;
  const int wm   = wid >> 1;      // 0..3 : which 32-row strip
  const int wn   = wid & 1;       // 0..1 : which 32-col strip
  const int half = lane >> 4;     // lane 16..31 half (ISA frag layouts)
  const int ln   = lane & 15;

  const int mBlock = blockIdx.y * 128;
  const int nBlock = blockIdx.x * 64;

  // staging coordinates
  const int srow = tid >> 1;      // 0..127 : x-tile row
  const int sseg = tid & 1;       // 16-elem segment within row
  const int wk   = tid >> 3;      // 0..31  : W-tile k row
  const int wnb  = tid & 7;       // 8-col group within W row

  v8f accG[4], accU[4];
#pragma unroll
  for (int i = 0; i < 4; ++i)
#pragma unroll
    for (int j = 0; j < 8; ++j) { accG[i][j] = 0.f; accU[i][j] = 0.f; }

  // in-flight W staging registers (issue -> commit); 8 bf16 each
  uint4 sg, su;

  // ---- issue: async-DMA X tile into LDS; start W b128 loads ----
  auto stage_issue = [&](int k0, int buf) {
    {
      const unsigned short* src = Xb + (size_t)(mBlock + srow) * K + k0 + sseg * 16;
      __builtin_prefetch(src + 32, 0, 1);            // speculative, next tile
      unsigned ldsa = (unsigned)(size_t)&xs[buf][srow * XS_STRIDE + sseg * 16];
      // INST_OFFSET applies to BOTH global and LDS address (ISA 10.x), so
      // offset:16 advances both sides; padded LDS layout preserved.
      asm volatile(
          "global_load_async_to_lds_b128 %0, %1, off\n\t"
          "global_load_async_to_lds_b128 %0, %1, off offset:16"
          :: "v"(ldsa), "v"(src) : "memory");
    }
    const unsigned short* srcg = Wb + (size_t)(k0 + wk) * NW + nBlock + wnb * 8;
    __builtin_prefetch(srcg + (size_t)32 * NW, 0, 1);
    sg = *reinterpret_cast<const uint4*>(srcg);
    if (GATED) su = *reinterpret_cast<const uint4*>(srcg + D_FF);  // "up" half
  };

  // ---- commit: scatter W rows into transposed LDS layout ----
  auto stage_commit = [&](int buf) {
    {
      unsigned v[4] = {sg.x, sg.y, sg.z, sg.w};
#pragma unroll
      for (int j = 0; j < 4; ++j) {
        wgT[buf][(wnb * 8 + 2 * j + 0) * WT_STRIDE + wk] = (unsigned short)(v[j]);
        wgT[buf][(wnb * 8 + 2 * j + 1) * WT_STRIDE + wk] = (unsigned short)(v[j] >> 16);
      }
    }
    if (GATED) {
      unsigned v[4] = {su.x, su.y, su.z, su.w};
#pragma unroll
      for (int j = 0; j < 4; ++j) {
        wuT[buf][(wnb * 8 + 2 * j + 0) * WT_STRIDE + wk] = (unsigned short)(v[j]);
        wuT[buf][(wnb * 8 + 2 * j + 1) * WT_STRIDE + wk] = (unsigned short)(v[j] >> 16);
      }
    }
  };

  // ---- pipeline prologue: tile 0 into buffer 0 ----
  stage_issue(0, 0);
  stage_commit(0);
  asm volatile("s_wait_asynccnt 0x0" ::: "memory");
  __syncthreads();

  const int NIT = K / 32;
  for (int it = 0; it < NIT; ++it) {
    const int  cur  = it & 1;
    const bool more = (it + 1 < NIT);
    if (more) stage_issue((it + 1) * 32, cur ^ 1);

    // ---- A fragments: 16x32 bf16, K runs {16*half..+15} ----
    Frag16 af[2];
#pragma unroll
    for (int rt = 0; rt < 2; ++rt) {
      const int row = wm * 32 + rt * 16 + ln;
      af[rt].q[0] = *reinterpret_cast<const uint4*>(&xs[cur][row * XS_STRIDE + 16 * half]);
      af[rt].q[1] = *reinterpret_cast<const uint4*>(&xs[cur][row * XS_STRIDE + 16 * half + 8]);
    }
    // ---- B fragments + WMMA ----
#pragma unroll
    for (int ct = 0; ct < 2; ++ct) {
      const int col = wn * 32 + ct * 16 + ln;
      Frag16 bg;
      const uint4* pg = reinterpret_cast<const uint4*>(&wgT[cur][col * WT_STRIDE + 16 * half]);
      bg.q[0] = pg[0];
      bg.q[1] = pg[1];
#pragma unroll
      for (int rt = 0; rt < 2; ++rt)
        accG[rt * 2 + ct] = __builtin_amdgcn_wmma_f32_16x16x32_bf16(
            false, af[rt].v, false, bg.v, (short)0, accG[rt * 2 + ct], false, false);
      if (GATED) {
        Frag16 bu;
        const uint4* pu = reinterpret_cast<const uint4*>(&wuT[cur][col * WT_STRIDE + 16 * half]);
        bu.q[0] = pu[0];
        bu.q[1] = pu[1];
#pragma unroll
        for (int rt = 0; rt < 2; ++rt)
          accU[rt * 2 + ct] = __builtin_amdgcn_wmma_f32_16x16x32_bf16(
              false, af[rt].v, false, bu.v, (short)0, accU[rt * 2 + ct], false, false);
      }
    }

    if (more) stage_commit(cur ^ 1);
    asm volatile("s_wait_asynccnt 0x0" ::: "memory");
    __syncthreads();    // next buffer published; cur free for overwrite next it
  }

  // ---- rank-16 LoRA epilogue: one extra zero-padded K=32 WMMA round ----
  Frag16 aL[2];
#pragma unroll
  for (int rt = 0; rt < 2; ++rt) {
    const float* xa = XA + (size_t)(mBlock + wm * 32 + rt * 16 + ln) * RANK;
    float4 f0 = reinterpret_cast<const float4*>(xa)[0];
    float4 f1 = reinterpret_cast<const float4*>(xa)[1];
    float4 f2 = reinterpret_cast<const float4*>(xa)[2];
    float4 f3 = reinterpret_cast<const float4*>(xa)[3];
    aL[rt].q[0] = make_uint4(bfpack2(f0.x, f0.y), bfpack2(f0.z, f0.w),
                             bfpack2(f1.x, f1.y), bfpack2(f1.z, f1.w));
    aL[rt].q[1] = make_uint4(bfpack2(f2.x, f2.y), bfpack2(f2.z, f2.w),
                             bfpack2(f3.x, f3.y), bfpack2(f3.z, f3.w));
    if (half) { aL[rt].q[0] = make_uint4(0, 0, 0, 0); aL[rt].q[1] = make_uint4(0, 0, 0, 0); }  // lanes 16..31 hold K=16..31 -> zero
  }
#pragma unroll
  for (int ct = 0; ct < 2; ++ct) {
    const int col = nBlock + wn * 32 + ct * 16 + ln;
    Frag16 bLg;
    bLg.q[0] = make_uint4(0, 0, 0, 0);
    bLg.q[1] = make_uint4(0, 0, 0, 0);
    if (half == 0) {                              // lanes 16..31 hold K>=16 -> zero
      unsigned us[8];
#pragma unroll
      for (int e = 0; e < 16; e += 2)
        us[e >> 1] = bfpack2(BL[(size_t)e * NW + col], BL[(size_t)(e + 1) * NW + col]);
      bLg.q[0] = make_uint4(us[0], us[1], us[2], us[3]);
      bLg.q[1] = make_uint4(us[4], us[5], us[6], us[7]);
    }
#pragma unroll
    for (int rt = 0; rt < 2; ++rt)
      accG[rt * 2 + ct] = __builtin_amdgcn_wmma_f32_16x16x32_bf16(
          false, aL[rt].v, false, bLg.v, (short)0, accG[rt * 2 + ct], false, false);
    if (GATED) {
      Frag16 bLu;
      bLu.q[0] = make_uint4(0, 0, 0, 0);
      bLu.q[1] = make_uint4(0, 0, 0, 0);
      if (half == 0) {
        unsigned us[8];
#pragma unroll
        for (int e = 0; e < 16; e += 2)
          us[e >> 1] = bfpack2(BL[(size_t)e * NW + col + D_FF],
                               BL[(size_t)(e + 1) * NW + col + D_FF]);
        bLu.q[0] = make_uint4(us[0], us[1], us[2], us[3]);
        bLu.q[1] = make_uint4(us[4], us[5], us[6], us[7]);
      }
#pragma unroll
      for (int rt = 0; rt < 2; ++rt)
        accU[rt * 2 + ct] = __builtin_amdgcn_wmma_f32_16x16x32_bf16(
            false, aL[rt].v, false, bLu.v, (short)0, accU[rt * 2 + ct], false, false);
    }
  }

  // ---- store: GATED -> h = silu(gate)*up as bf16 ; else fp32 out ----
  unsigned short* Hb = (unsigned short*)Out_;
  float*          Of = (float*)Out_;
#pragma unroll
  for (int rt = 0; rt < 2; ++rt)
#pragma unroll
    for (int ct = 0; ct < 2; ++ct) {
      const int idx  = rt * 2 + ct;
      const int row0 = mBlock + wm * 32 + rt * 16 + 8 * half;
      const int col  = nBlock + wn * 32 + ct * 16 + ln;
#pragma unroll
      for (int r = 0; r < 8; ++r) {
        float g = accG[idx][r];
        if (GATED) {
          float u = accU[idx][r];
          float hval = (g / (1.f + __expf(-g))) * u;   // SiLU(gate) * up
          Hb[(size_t)(row0 + r) * NOUT + col] = bf1(hval);
        } else {
          Of[(size_t)(row0 + r) * NOUT + col] = g;
        }
      }
    }
}

// ---------------------------------------------------------------------------
extern "C" void kernel_launch(void* const* d_in, const int* in_sizes, int n_in,
                              void* d_out, int out_size, void* d_ws, size_t ws_size,
                              hipStream_t stream) {
  (void)in_sizes; (void)n_in; (void)out_size; (void)ws_size;

  const float* x    = (const float*)d_in[0];   // [T, D_MODEL]
  const float* W_gu = (const float*)d_in[1];   // [D_MODEL, 2*D_FF]
  const float* A_gu = (const float*)d_in[2];   // [D_MODEL, RANK]
  const float* B_gu = (const float*)d_in[3];   // [RANK, 2*D_FF]
  const float* W_d  = (const float*)d_in[4];   // [D_FF, D_MODEL]
  const float* A_d  = (const float*)d_in[5];   // [D_FF, RANK]
  const float* B_d  = (const float*)d_in[6];   // [RANK, D_MODEL]

  // ws layout: XA f32 | HA f32 | H bf16 | x bf16 | W_gu bf16 | W_d bf16
  float*          XA   = (float*)d_ws;
  float*          HA   = XA + (size_t)TOKENS * RANK;
  unsigned short* H    = (unsigned short*)(HA + (size_t)TOKENS * RANK);
  unsigned short* Xbf  = H + (size_t)TOKENS * D_FF;
  unsigned short* Wgub = Xbf + (size_t)TOKENS * D_MODEL;
  unsigned short* Wdb  = Wgub + (size_t)D_MODEL * 2 * D_FF;

  const long nx  = (long)TOKENS * D_MODEL;     // 16,777,216
  const long nwg = (long)D_MODEL * 2 * D_FF;   //  5,767,168
  const long nwd = (long)D_FF * D_MODEL;       //  2,883,584

  // 0) one-shot bf16 conversion of x and weights (bandwidth-bound)
  cvt_f32_to_bf16<<<(unsigned)(nx / 2048), 256, 0, stream>>>(x, Xbf, nx);
  cvt_f32_to_bf16<<<(unsigned)(nwg / 2048), 256, 0, stream>>>(W_gu, Wgub, nwg);
  cvt_f32_to_bf16<<<(unsigned)(nwd / 2048), 256, 0, stream>>>(W_d, Wdb, nwd);

  // 1) XA = x @ A_gate_up
  rank_proj_f32<<<TOKENS / 16, 256, 0, stream>>>(x, A_gu, XA, D_MODEL);

  // 2) h = SiluAndMul(x @ W_gu + XA @ B_gu)   (bf16 out)
  fused_lora_gemm<1><<<dim3(D_FF / 64, TOKENS / 128), 256, 0, stream>>>(
      Xbf, Wgub, XA, B_gu, (void*)H, D_MODEL, 2 * D_FF, D_FF);

  // 3) HA = h @ A_down
  rank_proj_bf16<<<TOKENS / 16, 256, 0, stream>>>(H, A_d, HA, D_FF);

  // 4) out = h @ W_down + HA @ B_down        (fp32 out)
  fused_lora_gemm<0><<<dim3(D_MODEL / 64, TOKENS / 128), 256, 0, stream>>>(
      H, Wdb, HA, B_d, d_out, D_FF, D_MODEL, D_MODEL);
}
